// MultiQueryAttention_85907935855421
// MI455X (gfx1250) — compile-verified, matrix operand order checked by reading, not run on hardware
//
#include <hip/hip_runtime.h>
#include <hip/hip_bf16.h>
#include <stdint.h>

typedef __attribute__((ext_vector_type(16))) __bf16 v16bf;
typedef __attribute__((ext_vector_type(8)))  float  v8f;

union Frag {
    unsigned int   u[8];
    unsigned short us[16];
    v16bf          v;
};

// K-offset mapping for 16-bit A/B WMMA fragments (wave32):
// VGPR r, lane-half hf -> base K of the bf16 pair held in that dword.
__device__ __forceinline__ int koff(int r, int hf) {
    return ((r & 4) << 2) + hf * 8 + ((r & 3) << 1);
}

__device__ __forceinline__ unsigned short f2bf(float f) {
    unsigned int u = __float_as_uint(f);
    u += 0x7FFFu + ((u >> 16) & 1u);   // round-to-nearest-even
    return (unsigned short)(u >> 16);
}

// ---------------------------------------------------------------------------
// f32 -> bf16 elementwise convert
// ---------------------------------------------------------------------------
__global__ void cvt_f32_bf16(const float* __restrict__ s,
                             unsigned short* __restrict__ d, size_t n) {
    size_t i = (size_t)blockIdx.x * blockDim.x + threadIdx.x;
    if (i < n) d[i] = f2bf(s[i]);
}

// ---------------------------------------------------------------------------
// v (b, s, 256) f32 -> vT (b, 256, S) bf16 transpose (coalesced read)
// ---------------------------------------------------------------------------
__global__ void transpose_v(const float* __restrict__ src,
                            unsigned short* __restrict__ dst, int S) {
    const int Dh = 256;
    size_t i = (size_t)blockIdx.x * blockDim.x + threadIdx.x;   // over B*S*256
    const size_t perB = (size_t)S * Dh;
    const int b = (int)(i / perB);
    const size_t rem = i - (size_t)b * perB;
    const int s = (int)(rem / Dh);
    const int d = (int)(rem - (size_t)s * Dh);
    dst[((size_t)b * Dh + d) * S + s] = f2bf(src[i]);
}

// ---------------------------------------------------------------------------
// Tiled bf16 WMMA GEMM: C[M,N] = A[M,K] * B[K,N], A/B row-major bf16, C f32.
// 64x128 block tile, 8 waves, each wave: 1 M-tile x 4 N-tiles (A-frag reuse).
// A tile staged memory->LDS with GLOBAL_LOAD_ASYNC_TO_LDS_B128 (ASYNCcnt).
// Requires M%64==0, N%128==0, K%32==0 (true for all shapes here).
// ---------------------------------------------------------------------------
__global__ __launch_bounds__(256) void gemm_bf16(
        const unsigned short* __restrict__ A,
        const unsigned short* __restrict__ Bm,
        float* __restrict__ C, int M, int N, int K) {
    __shared__ unsigned short As[64][34];    // [m][k], padded
    __shared__ unsigned short Bs[128][34];   // [n][k] (transposed), padded

    const int tid  = threadIdx.x;
    const int lane = tid & 31;
    const int wave = tid >> 5;
    const int l16  = lane & 15;
    const int hf   = lane >> 4;
    const int m0   = blockIdx.y * 64;
    const int n0   = blockIdx.x * 128;
    const int mt   = wave >> 1;          // 0..3
    const int ntb  = (wave & 1) * 4;     // 0 or 4

    v8f acc[4];
    #pragma unroll
    for (int t = 0; t < 4; t++) acc[t] = (v8f){};

    const int arow = tid >> 2, acg = tid & 3;   // A staging coords
    const int bkr  = tid >> 3, bcg = tid & 7;   // B staging coords

    // per-lane LDS byte offset for the async A-tile DMA (low 32 bits of the
    // generic pointer to LDS are the LDS aperture offset)
    const unsigned ldsA = (unsigned)(uintptr_t)&As[arow][acg * 8];

    for (int k0 = 0; k0 < K; k0 += 32) {
        {   // stage A tile: 64 rows x 32 cols, async DMA memory -> LDS
            const unsigned short* gA = A + (size_t)(m0 + arow) * K + k0 + acg * 8;
            asm volatile("global_load_async_to_lds_b128 %0, %1, off"
                         :: "v"(ldsA), "v"(gA) : "memory");
        }
        #pragma unroll
        for (int hb = 0; hb < 2; hb++) {   // stage B tile transposed: 32 k x 128 n
            const int col = bcg * 8 + hb * 64;
            const uint4 vb = *(const uint4*)(Bm + (size_t)(k0 + bkr) * N + n0 + col);
            const unsigned short* e = (const unsigned short*)&vb;
            #pragma unroll
            for (int q = 0; q < 8; q++) Bs[col + q][bkr] = e[q];
        }
        if (k0 + 32 < K) {   // prefetch next K-step tiles
            __builtin_prefetch(A + (size_t)(m0 + arow) * K + (k0 + 32) + acg * 8, 0, 1);
            __builtin_prefetch(Bm + (size_t)(k0 + 32 + bkr) * N + n0 + bcg * 8, 0, 1);
        }
        asm volatile("s_wait_asynccnt 0" ::: "memory");   // A DMA complete
        __syncthreads();

        Frag af, bf[4];
        #pragma unroll
        for (int r = 0; r < 8; r++) {
            const int ko = koff(r, hf);
            af.u[r] = *(const unsigned int*)&As[mt * 16 + l16][ko];
            #pragma unroll
            for (int t = 0; t < 4; t++)
                bf[t].u[r] = *(const unsigned int*)&Bs[(ntb + t) * 16 + l16][ko];
        }
        #pragma unroll
        for (int t = 0; t < 4; t++)
            acc[t] = __builtin_amdgcn_wmma_f32_16x16x32_bf16(false, af.v, false, bf[t].v,
                                                             (short)0, acc[t], false, false);
        __syncthreads();
    }

    #pragma unroll
    for (int r = 0; r < 8; r++) {
        const int m = m0 + mt * 16 + r + 8 * hf;
        #pragma unroll
        for (int t = 0; t < 4; t++)
            C[(size_t)m * N + n0 + (ntb + t) * 16 + l16] = acc[t][r];
    }
}

// ---------------------------------------------------------------------------
// Fused RMSNorm (over D=256 per head) + interleaved RoPE, f32 in -> bf16 out.
// One block per (b,s,h); 128 threads, each owning one (even,odd) pair.
// ---------------------------------------------------------------------------
__global__ __launch_bounds__(128) void normrope(
        const float* __restrict__ src, unsigned short* __restrict__ dst,
        const float* __restrict__ w,
        const float* __restrict__ cosb, const float* __restrict__ sinb,
        int S, int Hn) {
    const int t = threadIdx.x;
    const long long blk = blockIdx.x;
    const long long bs  = blk / Hn;
    const int s = (int)(bs % S);
    const size_t base = (size_t)blk * 256;

    const float a = src[base + 2 * t];
    const float b = src[base + 2 * t + 1];

    __shared__ float red[128];
    red[t] = a * a + b * b;
    __syncthreads();
    #pragma unroll
    for (int o = 64; o > 0; o >>= 1) {
        if (t < o) red[t] += red[t + o];
        __syncthreads();
    }
    const float inv = rsqrtf(red[0] * (1.0f / 256.0f) + 1e-6f);
    const float na = a * inv * (1.0f + w[2 * t]);
    const float nb = b * inv * (1.0f + w[2 * t + 1]);
    const float cv = cosb[(size_t)s * 128 + t];
    const float sv = sinb[(size_t)s * 128 + t];
    dst[base + 2 * t]     = f2bf(na * cv - nb * sv);
    dst[base + 2 * t + 1] = f2bf(na * sv + nb * cv);
}

// ---------------------------------------------------------------------------
// Sliding-window flash attention, one wave per 16-query tile.
// q: (b,s,h,256) bf16   k: (b,s,256) bf16   vT: (b,256,S) bf16
// out: (b,s,h,256) bf16
// Note: jb is 32-aligned and jb <= q0+15 <= S-1  =>  jb+31 <= S-1 (no OOB).
// ---------------------------------------------------------------------------
__global__ __launch_bounds__(32) void mqa_flash(
        const unsigned short* __restrict__ qbf,
        const unsigned short* __restrict__ kbf,
        const unsigned short* __restrict__ vT,
        unsigned short* __restrict__ obf, int S) {
    const int Hn = 4, Dh = 256, WIN = 512;
    const float SCALE = 0.0625f;                 // 256^-0.5
    const float NEGINF = __builtin_inff() * -1.0f;

    __shared__ unsigned short P[16][34];         // p-tile re-layout buffer

    const int lane = threadIdx.x;
    const int l16  = lane & 15;
    const int hf   = lane >> 4;
    const int q0   = blockIdx.x * 16;
    const int h    = blockIdx.y;
    const int b    = blockIdx.z;

    // resident Q fragments: 16 rows x 256 d, 8 chunks of K=32
    const size_t qrow = (((size_t)b * S + q0 + l16) * Hn + h) * Dh;
    Frag qf[8];
    #pragma unroll
    for (int c = 0; c < 8; c++)
        #pragma unroll
        for (int r = 0; r < 8; r++)
            qf[c].u[r] = *(const unsigned int*)(qbf + qrow + c * 32 + koff(r, hf));

    float mi[8], li[8];
    #pragma unroll
    for (int r = 0; r < 8; r++) { mi[r] = NEGINF; li[r] = 0.0f; }
    v8f o[16];
    #pragma unroll
    for (int t = 0; t < 16; t++) o[t] = (v8f){};

    const size_t kbase = (size_t)b * S * Dh;
    const size_t vbase = (size_t)b * Dh * S;
    const int jstart = (q0 > WIN) ? ((q0 - WIN) & ~31) : 0;

    for (int jb = jstart; jb <= q0 + 15; jb += 32) {
        if (jb + 32 <= q0 + 15)   // prefetch next key block rows
            __builtin_prefetch(kbf + kbase + (size_t)(jb + 32 + l16) * Dh, 0, 1);

        // ---- scores: Q(16x256) @ K[jb..jb+31]^T -> two 16x16 tiles ----
        v8f s0 = {}, s1 = {};
        const size_t kr0 = kbase + (size_t)(jb + l16) * Dh;
        const size_t kr1 = kbase + (size_t)(jb + 16 + l16) * Dh;
        #pragma unroll
        for (int c = 0; c < 8; c++) {
            Frag k0f, k1f;
            #pragma unroll
            for (int r = 0; r < 8; r++) {
                const int d = c * 32 + koff(r, hf);
                k0f.u[r] = *(const unsigned int*)(kbf + kr0 + d);
                k1f.u[r] = *(const unsigned int*)(kbf + kr1 + d);
            }
            s0 = __builtin_amdgcn_wmma_f32_16x16x32_bf16(false, qf[c].v, false, k0f.v,
                                                         (short)0, s0, false, false);
            s1 = __builtin_amdgcn_wmma_f32_16x16x32_bf16(false, qf[c].v, false, k1f.v,
                                                         (short)0, s1, false, false);
        }

        // ---- mask + online softmax (rows live across one 16-lane half) ----
        float alpha[8];
        #pragma unroll
        for (int r = 0; r < 8; r++) {
            const int i  = q0 + r + 8 * hf;
            const int j0 = jb + l16, j1 = j0 + 16;
            float v0 = (j0 <= i && (i - j0) <= WIN) ? s0[r] * SCALE : NEGINF;
            float v1 = (j1 <= i && (i - j1) <= WIN) ? s1[r] * SCALE : NEGINF;
            float mx = fmaxf(v0, v1);
            #pragma unroll
            for (int ofs = 8; ofs > 0; ofs >>= 1)
                mx = fmaxf(mx, __shfl_xor(mx, ofs, 16));
            const float mtot  = fmaxf(mi[r], mx);
            const float msafe = (mtot == NEGINF) ? 0.0f : mtot;
            const float al    = (mi[r] == NEGINF) ? 0.0f : __expf(mi[r] - msafe);
            const float p0 = (v0 == NEGINF) ? 0.0f : __expf(v0 - msafe);
            const float p1 = (v1 == NEGINF) ? 0.0f : __expf(v1 - msafe);
            float rs = p0 + p1;
            #pragma unroll
            for (int ofs = 8; ofs > 0; ofs >>= 1)
                rs += __shfl_xor(rs, ofs, 16);
            li[r] = li[r] * al + rs;
            mi[r] = mtot;
            alpha[r] = al;
            s0[r] = p0; s1[r] = p1;
        }
        #pragma unroll
        for (int t = 0; t < 16; t++)
            #pragma unroll
            for (int r = 0; r < 8; r++) o[t][r] *= alpha[r];

        // ---- re-layout P (C-layout f32 -> A-fragment bf16) via LDS ----
        __syncthreads();
        #pragma unroll
        for (int r = 0; r < 8; r++) {
            const int row = r + 8 * hf;
            P[row][l16]      = f2bf(s0[r]);
            P[row][16 + l16] = f2bf(s1[r]);
        }
        __syncthreads();
        Frag pf;
        #pragma unroll
        for (int r = 0; r < 8; r++)
            pf.u[r] = *(const unsigned int*)&P[l16][koff(r, hf)];

        // ---- O += P(16x32) @ V(32x256): V^T rows give contiguous K-pairs ----
        #pragma unroll
        for (int t = 0; t < 16; t++) {
            const size_t vrow = vbase + (size_t)(t * 16 + l16) * S + jb;
            Frag vf;
            #pragma unroll
            for (int r = 0; r < 8; r++)
                vf.u[r] = *(const unsigned int*)(vT + vrow + koff(r, hf));
            o[t] = __builtin_amdgcn_wmma_f32_16x16x32_bf16(false, pf.v, false, vf.v,
                                                           (short)0, o[t], false, false);
        }
    }

    // ---- epilogue: scale by 1/l and store bf16 ----
    #pragma unroll
    for (int r = 0; r < 8; r++) {
        const int row = r + 8 * hf;
        const float inv = 1.0f / li[r];
        const size_t ob = (((size_t)b * S + q0 + row) * Hn + h) * Dh;
        #pragma unroll
        for (int t = 0; t < 16; t++)
            obf[ob + t * 16 + l16] = f2bf(o[t][r] * inv);
    }
}

// ---------------------------------------------------------------------------
// Host-side orchestration
// ---------------------------------------------------------------------------
extern "C" void kernel_launch(void* const* d_in, const int* in_sizes, int n_in,
                              void* d_out, int out_size, void* d_ws, size_t ws_size,
                              hipStream_t stream) {
    const float* x    = (const float*)d_in[0];
    const float* cosb = (const float*)d_in[1];
    const float* sinb = (const float*)d_in[2];
    const float* Wq   = (const float*)d_in[3];
    const float* Wk   = (const float*)d_in[4];
    const float* Wv   = (const float*)d_in[5];
    const float* Wo   = (const float*)d_in[6];
    const float* qw   = (const float*)d_in[7];
    const float* kw   = (const float*)d_in[8];
    float* out = (float*)d_out;

    const int Bb = 2, S = 4096, E = 640, H = 4, D = 256, HD = 1024;
    const size_t M = (size_t)Bb * S;   // 8192 token rows

    char* ws = (char*)d_ws;
    size_t off = 0;
    auto take = [&](size_t bytes) -> char* {
        char* p = ws + off;
        off += (bytes + 255) & ~(size_t)255;
        return p;
    };
    unsigned short* xbf  = (unsigned short*)take(M * E * 2);
    unsigned short* wqbf = (unsigned short*)take((size_t)E * HD * 2);
    unsigned short* wkbf = (unsigned short*)take((size_t)E * D * 2);
    unsigned short* wvbf = (unsigned short*)take((size_t)E * D * 2);
    unsigned short* wobf = (unsigned short*)take((size_t)HD * E * 2);
    float* q_raw = (float*)take(M * HD * 4);
    float* k_raw = (float*)take(M * D * 4);
    float* v_raw = (float*)take(M * D * 4);
    unsigned short* qbf = (unsigned short*)take(M * HD * 2);
    unsigned short* kbf = (unsigned short*)take(M * D * 2);
    unsigned short* vT  = (unsigned short*)take(M * D * 2 + 256);  // +tail pad
    unsigned short* abf = (unsigned short*)q_raw;   // reuse q_raw for attn out

    auto cvt = [&](const float* s, unsigned short* d, size_t n) {
        const int thr = 256;
        cvt_f32_bf16<<<dim3((unsigned)((n + thr - 1) / thr)), dim3(thr), 0, stream>>>(s, d, n);
    };

    // 1) convert activations + weights to bf16
    cvt(x,  xbf,  M * E);
    cvt(Wq, wqbf, (size_t)E * HD);
    cvt(Wk, wkbf, (size_t)E * D);
    cvt(Wv, wvbf, (size_t)E * D);
    cvt(Wo, wobf, (size_t)HD * E);

    // 2) QKV projections (WMMA GEMMs), 64x128 block tiles
    gemm_bf16<<<dim3(HD / 128, (unsigned)(M / 64)), dim3(256), 0, stream>>>(
        xbf, wqbf, q_raw, (int)M, HD, E);
    gemm_bf16<<<dim3(D / 128, (unsigned)(M / 64)), dim3(256), 0, stream>>>(
        xbf, wkbf, k_raw, (int)M, D, E);
    gemm_bf16<<<dim3(D / 128, (unsigned)(M / 64)), dim3(256), 0, stream>>>(
        xbf, wvbf, v_raw, (int)M, D, E);

    // 3) RMSNorm + RoPE -> bf16; V transposed to (b, d, s) bf16
    normrope<<<dim3((unsigned)(M * H)), dim3(128), 0, stream>>>(
        q_raw, qbf, qw, cosb, sinb, S, H);
    normrope<<<dim3((unsigned)M), dim3(128), 0, stream>>>(
        k_raw, kbf, kw, cosb, sinb, S, 1);
    transpose_v<<<dim3((unsigned)(M * D / 256)), dim3(256), 0, stream>>>(v_raw, vT, S);

    // 4) sliding-window flash attention (WMMA)
    mqa_flash<<<dim3(S / 16, H, Bb), dim3(32), 0, stream>>>(qbf, kbf, vT, abf, S);

    // 5) output projection (WMMA GEMM) -> f32 out
    gemm_bf16<<<dim3(E / 128, (unsigned)(M / 64)), dim3(256), 0, stream>>>(
        abf, wobf, out, (int)M, E, HD);

    (void)in_sizes; (void)n_in; (void)out_size; (void)ws_size;
}
